// SpectPixelPieNeRF_62981400429157
// MI455X (gfx1250) — matrix operand order repaired and checked by reading, not compiled, not run on hardware
//
#include <hip/hip_runtime.h>
#include <hip/hip_bf16.h>

typedef __attribute__((ext_vector_type(16))) _Float16 v16h;
typedef __attribute__((ext_vector_type(8)))  _Float16 v8h;
typedef __attribute__((ext_vector_type(8)))  float    v8f;

#define NPTS   (64*64*64)   // 262144 points per view
#define WID    256
#define INP_K  576          // 551 padded to multiple of 32
#define INP_S  584          // LDS row stride for inp (bank-friendly)
#define L5_K   832          // 807 padded; h part mapped to k in [576,832)
#define H_S    264          // LDS row stride for hidden activations
#define MTILE  128
#define BLOCK  512          // 16 wave32s

// Input vector layout (reordered for aligned vector fills):
//   cols [0,512)   : latent z
//   cols [512,551) : positional encoding (39)
//   cols [551,576) : zero pad
// Weight rows are remapped to match during conversion.

// ---------------------------------------------------------------------------
// Weight conversion: f32 [fanIn][256] -> f16 transposed [256][Kpad], zero pad.
// mode 0: identity rows (hidden layers)
// mode 1: inp-reordered rows:  k<512 -> 39+k ; 512<=k<551 -> k-512 ;
//         551<=k<576 -> zero ; k>=576 -> k-25   (covers layer 0 and layer 5)
// ---------------------------------------------------------------------------
__global__ void k_convert_w(const float* __restrict__ src, _Float16* __restrict__ dst,
                            int fanIn, int Kpad, int mode, int total) {
    int t = blockIdx.x * blockDim.x + threadIdx.x;
    if (t >= total) return;
    int col = t / Kpad;
    int k   = t - col * Kpad;
    int sk;
    if (mode == 0) {
        sk = k;
    } else {
        if      (k < 512) sk = 39 + k;
        else if (k < 551) sk = k - 512;
        else if (k < 576) sk = -1;
        else              sk = k - 25;    // layer-5 h part: 551..806
    }
    float v = 0.0f;
    if (sk >= 0 && sk < fanIn) v = src[sk * WID + col];
    dst[col * Kpad + k] = (_Float16)v;
}

__global__ void k_convert_vec(const float* __restrict__ src, _Float16* __restrict__ dst, int n) {
    int t = blockIdx.x * blockDim.x + threadIdx.x;
    if (t < n) dst[t] = (_Float16)src[t];
}

// ---------------------------------------------------------------------------
// Encoder feature map: feat[3][64][64] = { mean_d ct, ap, pa }
// ---------------------------------------------------------------------------
__global__ void k_feat(const float* __restrict__ ct, const float* __restrict__ ap,
                       const float* __restrict__ pa, float* __restrict__ feat) {
    int idx = blockIdx.x * blockDim.x + threadIdx.x;
    if (idx >= 3 * 4096) return;
    int ch = idx >> 12;
    int p  = idx & 4095;          // h*64 + w
    float v;
    if (ch == 0) {
        float s = 0.0f;
        for (int d = 0; d < 64; ++d) s += ct[(d << 12) + p];
        v = s * (1.0f / 64.0f);
    } else if (ch == 1) v = ap[p];
    else                v = pa[p];
    feat[idx] = v;
}

// conv 3->64, 3x3, stride 2, SAME (pad_lo=0, pad_hi=1), relu
__global__ void k_conv1(const float* __restrict__ feat, const float* __restrict__ w,
                        const float* __restrict__ b, float* __restrict__ out) {
    int idx = blockIdx.x * blockDim.x + threadIdx.x;
    if (idx >= 64 * 32 * 32) return;
    int o  = idx >> 10;
    int oh = (idx >> 5) & 31;
    int ow = idx & 31;
    float acc = b[o];
    for (int i = 0; i < 3; ++i)
        for (int kh = 0; kh < 3; ++kh) {
            int ih = oh * 2 + kh; if (ih > 63) continue;
            for (int kw = 0; kw < 3; ++kw) {
                int iw = ow * 2 + kw; if (iw > 63) continue;
                acc += feat[(i << 12) + (ih << 6) + iw] * w[((o * 3 + i) * 3 + kh) * 3 + kw];
            }
        }
    out[idx] = acc > 0.0f ? acc : 0.0f;
}

// conv 64->128, 3x3, stride 2, SAME, relu
__global__ void k_conv2(const float* __restrict__ in, const float* __restrict__ w,
                        const float* __restrict__ b, float* __restrict__ out) {
    int idx = blockIdx.x * blockDim.x + threadIdx.x;
    if (idx >= 128 * 16 * 16) return;
    int o  = idx >> 8;
    int oh = (idx >> 4) & 15;
    int ow = idx & 15;
    float acc = b[o];
    for (int i = 0; i < 64; ++i)
        for (int kh = 0; kh < 3; ++kh) {
            int ih = oh * 2 + kh; if (ih > 31) continue;
            for (int kw = 0; kw < 3; ++kw) {
                int iw = ow * 2 + kw; if (iw > 31) continue;
                acc += in[(i << 10) + (ih << 5) + iw] * w[((o * 64 + i) * 3 + kh) * 3 + kw];
            }
        }
    out[idx] = acc > 0.0f ? acc : 0.0f;
}

// global mean pool (128 ch over 16x16) then FC 128->512, emit z as f16
__global__ void k_poolfc(const float* __restrict__ c2, const float* __restrict__ fcw,
                         const float* __restrict__ fcb, _Float16* __restrict__ zh) {
    __shared__ float g[128];
    int t = threadIdx.x;
    if (t < 128) {
        float s = 0.0f;
        for (int i = 0; i < 256; ++i) s += c2[(t << 8) + i];
        g[t] = s * (1.0f / 256.0f);
    }
    __syncthreads();
    if (t < 512) {
        float z = fcb[t];
        for (int c = 0; c < 128; ++c) z += g[c] * fcw[c * 512 + t];
        zh[t] = (_Float16)z;
    }
}

// ---------------------------------------------------------------------------
// WMMA fragment loaders (wave32); A and B share contiguous k = 16*(lane/16) + s
// ---------------------------------------------------------------------------
__device__ __forceinline__ v16h load_a_frag(const _Float16* src, int stride,
                                            int rowTile, int kbase, int lane) {
    // 16-bit A 16x32: lane<16 -> row=lane, K kbase+0..15
    //                 lane>=16 -> row=lane-16, K kbase+16..31
    int row = rowTile * 16 + (lane & 15);
    const _Float16* p = src + row * stride + kbase + ((lane >> 4) << 4);
    v8h lo = *(const v8h*)p;
    v8h hi = *(const v8h*)(p + 8);
    v16h a;
#pragma unroll
    for (int i = 0; i < 8; ++i) { a[i] = lo[i]; a[i + 8] = hi[i]; }
    return a;
}

__device__ __forceinline__ v16h load_b_frag(const _Float16* wt, int Kpad,
                                            int colTile, int kstep, int lane) {
    // 16-bit B 32x16: lane<16 -> col=lane, K kstep*32+0..15
    //                 lane>=16 -> col=lane-16, K kstep*32+16..31
    int col = colTile * 16 + (lane & 15);
    int kb  = kstep * 32 + ((lane >> 4) << 4);
    return *(const v16h*)(wt + col * Kpad + kb);
}

__device__ __forceinline__ void gemm_seg(const _Float16* src, int stride, int ksteps,
                                         const _Float16* wt, int Kpad, int kstep0,
                                         int rt, int ch, int lane, v8f acc[8]) {
    for (int ks = 0; ks < ksteps; ++ks) {
        v16h a = load_a_frag(src, stride, rt, ks * 32, lane);
#pragma unroll
        for (int j = 0; j < 8; ++j) {
            v16h b = load_b_frag(wt, Kpad, ch * 8 + j, kstep0 + ks, lane);
            acc[j] = __builtin_amdgcn_wmma_f32_16x16x32_f16(
                false, a, false, b, (short)0, acc[j], false, false);
        }
    }
}

__device__ __forceinline__ void epilogue_relu(v8f acc[8], const float* bias,
                                              _Float16* dst, int rt, int ch, int lane) {
    // C/D 16x16 f32: lane<16 -> N=lane, M=vgpr; lane>=16 -> N=lane-16, M=8+vgpr
    int rowbase = rt * 16 + ((lane >> 4) << 3);
#pragma unroll
    for (int j = 0; j < 8; ++j) {
        int col = (ch * 8 + j) * 16 + (lane & 15);
        float bj = bias[col];
#pragma unroll
        for (int r = 0; r < 8; ++r) {
            float v = acc[j][r] + bj;
            v = v > 0.0f ? v : 0.0f;
            dst[(rowbase + r) * H_S + col] = (_Float16)v;
        }
    }
}

struct MlpArgs {
    const _Float16* wt[8];
    const float*    bias[8];
    const _Float16* outw;
    const float*    outb;
    const _Float16* zh;
    float*          sig;     // [2*NPTS] flat (AP then PA)
};

__global__ __launch_bounds__(BLOCK)
void k_mlp(MlpArgs args) {
    extern __shared__ _Float16 lds[];
    _Float16* inp = lds;                         // MTILE x INP_S
    _Float16* hb0 = lds + MTILE * INP_S;         // MTILE x H_S
    _Float16* hb1 = hb0 + MTILE * H_S;           // MTILE x H_S

    const int t    = threadIdx.x;
    const int lane = t & 31;
    const int wave = t >> 5;                 // 0..15
    const int rt   = wave >> 1;              // row tile 0..7
    const int ch   = wave & 1;               // column half 0..1

    const int g0   = blockIdx.x * MTILE;     // flat over 2*NPTS
    const int view = (g0 >= NPTS) ? 1 : 0;
    const int n0   = g0 - view * NPTS;

    // --- fill latent part of inp: cols [0,512), 16B-vectorized ---
    for (int idx = t; idx < MTILE * 64; idx += BLOCK) {       // 64 v8h chunks/row
        int r = idx >> 6, c = idx & 63;
        *(v8h*)(inp + r * INP_S + c * 8) = *(const v8h*)(args.zh + c * 8);
    }
    // --- positional encoding: cols [512,551), zero pad [551,576) ---
    if (t < MTILE) {
        int n  = n0 + t;
        int di = n >> 12;
        int wi = (n >> 6) & 63;
        int hi = n & 63;
        float x = (float)wi - 31.5f;
        float y = view ? (31.5f - (float)hi) : ((float)hi - 31.5f);
        float z = (float)di - 31.5f;
        _Float16* row = inp + t * INP_S + 512;
        float v3[3] = {x, y, z};
        row[0] = (_Float16)x; row[1] = (_Float16)y; row[2] = (_Float16)z;
#pragma unroll
        for (int a = 0; a < 3; ++a) {
            float f = v3[a];
#pragma unroll
            for (int l = 0; l < 6; ++l) {
                float arg = f * (float)(1 << l);
                row[3  + a * 6 + l] = (_Float16)sinf(arg);
                row[21 + a * 6 + l] = (_Float16)cosf(arg);
            }
        }
        for (int k = 39; k < 64; ++k) row[k] = (_Float16)0.0f;   // cols 551..575
    }
    __syncthreads();

    v8f zf = {0.f, 0.f, 0.f, 0.f, 0.f, 0.f, 0.f, 0.f};

    // layer 0: [MTILE x 576] x [576 x 256]
    {
        v8f acc[8];
#pragma unroll
        for (int j = 0; j < 8; ++j) acc[j] = zf;
        gemm_seg(inp, INP_S, INP_K / 32, args.wt[0], INP_K, 0, rt, ch, lane, acc);
        epilogue_relu(acc, args.bias[0], hb0, rt, ch, lane);
    }
    __syncthreads();

    _Float16* cur = hb0;
    _Float16* nxt = hb1;
    for (int i = 1; i < 8; ++i) {
        v8f acc[8];
#pragma unroll
        for (int j = 0; j < 8; ++j) acc[j] = zf;
        if (i == 5) {
            // skip concat: [inp | h] x wt5 (K padded 832, h rows at k>=576)
            gemm_seg(inp, INP_S, INP_K / 32, args.wt[5], L5_K, 0,          rt, ch, lane, acc);
            gemm_seg(cur, H_S,   WID / 32,   args.wt[5], L5_K, INP_K / 32, rt, ch, lane, acc);
        } else {
            gemm_seg(cur, H_S, WID / 32, args.wt[i], WID, 0, rt, ch, lane, acc);
        }
        epilogue_relu(acc, args.bias[i], nxt, rt, ch, lane);
        __syncthreads();
        _Float16* tmp = cur; cur = nxt; nxt = tmp;
    }

    // output layer: h[256] . out_w + out_b  (cur == layer-7 activations)
    if (t < MTILE) {
        float s = *args.outb;
        const _Float16* hr = cur + t * H_S;
#pragma unroll
        for (int k = 0; k < WID; k += 8) {
            v8h hv = *(const v8h*)(hr + k);
            v8h wv = *(const v8h*)(args.outw + k);
#pragma unroll
            for (int e = 0; e < 8; ++e) s = fmaf((float)hv[e], (float)wv[e], s);
        }
        args.sig[g0 + t] = s;
    }
}

// ---------------------------------------------------------------------------
// Projection: sigma = 0.5*(sigA+sigP); ap/pa line integrals with exp(-cum mu)
// out: [ap 4096][pa 4096][sigma_vol 262144]
// ---------------------------------------------------------------------------
__global__ void k_spect(const float* __restrict__ sigA, const float* __restrict__ sigP,
                        const float* __restrict__ ct, float* __restrict__ out) {
    int idx = blockIdx.x * blockDim.x + threadIdx.x;
    if (idx >= 4096) return;
    int d = idx >> 6;
    int w = idx & 63;
    float tau = 0.0f, apv = 0.0f;
    for (int h = 0; h < 64; ++h) {
        int n = (((d << 6) + w) << 6) + h;                 // sigma (D,W,H) flat
        float s = 0.5f * (sigA[n] + sigP[n]);
        int vol = ((d << 6) + h) * 64 + w;                 // (D,H,W) flat
        out[8192 + vol] = s;                               // sigma_vol
        float c  = ct[vol];
        float mu = (c > 0.0f ? c : 0.0f) * 0.01f;
        apv += s * expf(-tau);
        tau += mu;                                         // STEP = 1
    }
    out[idx] = apv;
    tau = 0.0f;
    float pav = 0.0f;
    for (int h = 63; h >= 0; --h) {
        int n = (((d << 6) + w) << 6) + h;
        float s = 0.5f * (sigA[n] + sigP[n]);
        float c  = ct[((d << 6) + h) * 64 + w];
        float mu = (c > 0.0f ? c : 0.0f) * 0.01f;
        pav += s * expf(-tau);
        tau += mu;
    }
    out[4096 + idx] = pav;
}

// ---------------------------------------------------------------------------
extern "C" void kernel_launch(void* const* d_in, const int* in_sizes, int n_in,
                              void* d_out, int out_size, void* d_ws, size_t ws_size,
                              hipStream_t stream) {
    (void)n_in; (void)out_size; (void)ws_size;

    const float *ct, *ap, *pa, *enc_w1, *enc_b1, *enc_w2, *enc_b2;
    const float *fc_w, *fc_b, *out_w, *out_b;
    const float *mlp_w[8], *mlp_b[8];

    if (in_sizes[0] == NPTS) {
        // dict insertion order: ct, ap, pa, enc_w1, enc_b1, enc_w2, enc_b2,
        //                       fc_w, fc_b, out_w, out_b, (mlp_w, mlp_b)*8
        ct = (const float*)d_in[0];  ap = (const float*)d_in[1];  pa = (const float*)d_in[2];
        enc_w1 = (const float*)d_in[3];  enc_b1 = (const float*)d_in[4];
        enc_w2 = (const float*)d_in[5];  enc_b2 = (const float*)d_in[6];
        fc_w = (const float*)d_in[7];    fc_b = (const float*)d_in[8];
        out_w = (const float*)d_in[9];   out_b = (const float*)d_in[10];
        for (int i = 0; i < 8; ++i) {
            mlp_w[i] = (const float*)d_in[11 + 2 * i];
            mlp_b[i] = (const float*)d_in[12 + 2 * i];
        }
    } else {
        // alphabetically sorted flatten (jax pytree): ap, ct, pa, enc_b1, enc_b2,
        // enc_w1, enc_w2, fc_b, fc_w, mlp_b0..7, mlp_w0..7, out_b, out_w
        ap = (const float*)d_in[0];  ct = (const float*)d_in[1];  pa = (const float*)d_in[2];
        enc_b1 = (const float*)d_in[3];  enc_b2 = (const float*)d_in[4];
        enc_w1 = (const float*)d_in[5];  enc_w2 = (const float*)d_in[6];
        fc_b = (const float*)d_in[7];    fc_w = (const float*)d_in[8];
        for (int i = 0; i < 8; ++i) mlp_b[i] = (const float*)d_in[9 + i];
        for (int i = 0; i < 8; ++i) mlp_w[i] = (const float*)d_in[17 + i];
        out_b = (const float*)d_in[25];  out_w = (const float*)d_in[26];
    }

    // workspace carve
    size_t off = 0;
    auto carve = [&](size_t bytes) -> void* {
        void* p = (char*)d_ws + off;
        off += (bytes + 255) & ~(size_t)255;
        return p;
    };
    const int kpad[8]  = {INP_K, WID, WID, WID, WID, L5_K, WID, WID};
    const int fanin[8] = {551,   WID, WID, WID, WID, 807,  WID, WID};
    const int mode8[8] = {1,     0,   0,   0,   0,   1,    0,   0};
    _Float16* wt[8];
    for (int i = 0; i < 8; ++i) wt[i] = (_Float16*)carve((size_t)WID * kpad[i] * 2);
    _Float16* outw_h = (_Float16*)carve(WID * 2);
    _Float16* zh     = (_Float16*)carve(512 * 2);
    float* feat = (float*)carve((size_t)3 * 4096 * 4);
    float* c1   = (float*)carve((size_t)64 * 32 * 32 * 4);
    float* c2   = (float*)carve((size_t)128 * 16 * 16 * 4);
    float* sig  = (float*)carve((size_t)2 * NPTS * 4);   // AP then PA

    // 1) weight conversion (f32 -> transposed/padded/row-remapped f16)
    for (int i = 0; i < 8; ++i) {
        int total = WID * kpad[i];
        k_convert_w<<<(total + 255) / 256, 256, 0, stream>>>(
            mlp_w[i], wt[i], fanin[i], kpad[i], mode8[i], total);
    }
    k_convert_vec<<<1, 256, 0, stream>>>(out_w, outw_h, WID);

    // 2) encoder
    k_feat<<<(3 * 4096 + 255) / 256, 256, 0, stream>>>(ct, ap, pa, feat);
    k_conv1<<<(64 * 32 * 32 + 255) / 256, 256, 0, stream>>>(feat, enc_w1, enc_b1, c1);
    k_conv2<<<(128 * 16 * 16 + 255) / 256, 256, 0, stream>>>(c1, enc_w2, enc_b2, c2);
    k_poolfc<<<1, 512, 0, stream>>>(c2, fc_w, fc_b, zh);

    // 3) WMMA MLP over both views
    MlpArgs margs;
    for (int i = 0; i < 8; ++i) { margs.wt[i] = wt[i]; margs.bias[i] = mlp_b[i]; }
    margs.outw = outw_h; margs.outb = out_b; margs.zh = zh; margs.sig = sig;
    const int nblocks = (2 * NPTS) / MTILE;   // 4096
    const size_t lds_bytes = (size_t)(MTILE * INP_S + 2 * MTILE * H_S) * sizeof(_Float16);
    k_mlp<<<nblocks, BLOCK, lds_bytes, stream>>>(margs);

    // 4) projection + outputs
    k_spect<<<(4096 + 255) / 256, 256, 0, stream>>>(sig, sig + NPTS, ct, (float*)d_out);
}
